// MoEFFN_63178968924533
// MI455X (gfx1250) — matrix operand order repaired and checked by reading, not compiled, not run on hardware
//
#include <hip/hip_runtime.h>
#include <hip/hip_bf16.h>

#define NT    8192          // B*T tokens
#define DM    1024          // D_MODEL
#define DFF   4096          // D_FF
#define DHID  512           // D_HID
#define NEXP  4
#define CHUNK 2048          // tokens per expert chunk (4 chunks cover worst case 8192)
#define LDA   40            // LDS row stride in bf16 elems (32 + pad, keeps 16B row alignment)

typedef __attribute__((ext_vector_type(16))) __bf16 v16bf;
typedef __attribute__((ext_vector_type(8)))  float  v8f;

union Frag { v16bf v; unsigned int u[8]; };

__device__ __forceinline__ unsigned short f2bf(float f) {
  unsigned int u = __float_as_uint(f);
  u += 0x7FFFu + ((u >> 16) & 1u);          // round-to-nearest-even
  return (unsigned short)(u >> 16);
}
__device__ __forceinline__ unsigned int f2bf_pk(float lo, float hi) {
  return (unsigned int)f2bf(lo) | ((unsigned int)f2bf(hi) << 16);
}
__device__ __forceinline__ float bf2f(unsigned short s) {
  return __uint_as_float(((unsigned int)s) << 16);
}
__device__ __forceinline__ float gelu_f(float x) {
  return 0.5f * x * (1.0f + erff(x * 0.70710678118654752f));
}

// ---------------------------------------------------------------------------
// Core WMMA main loop: C[128x128] += A[128xK](bf16, optional row-gather) * B.
// BT=true : B pre-transposed bf16 [N][K] -> staging is pure b128 copies.
// BT=false: B fp32 row-major [K][N], converted/transposed on the LDS stage.
// 256 threads = 8 waves (wave32); wave w owns rows [16w,16w+16); 8 v8f accs.
// Software-pipelined: double-buffered LDS, one barrier per K-step; next
// tile's global loads and the next B fragment's LDS loads are both issued
// before the WMMA that depends on the previous ones.
// ---------------------------------------------------------------------------
template <bool BT>
__device__ __forceinline__ void mma_tile(
    const unsigned short* __restrict__ Abase, int Kdim,
    const int* __restrict__ rowmap, int rowbase, int nvalid,
    const float* __restrict__ Bf, const unsigned short* __restrict__ Bt,
    int Ndim, int n0,
    unsigned short (*ldsA)[128 * LDA], unsigned short (*ldsB)[128 * LDA],
    v8f acc[8])
{
  const int tid  = threadIdx.x;
  const int wave = tid >> 5;
  const int lane = tid & 31;
  const int lh   = lane & 15;
  const bool hi  = lane >= 16;

  // ---- hoisted per-thread staging coordinates (invariant over K) ----------
  const unsigned short* a_src[2]; int a_off[2]; bool a_ok[2];
  {
    int arow0 = rowmap ? rowmap[rowbase] : rowbase;   // safe clamp row
#pragma unroll
    for (int i = 0; i < 2; ++i) {
      int idx = tid + i * 256;
      int r = idx >> 2, kg = idx & 3;
      a_ok[i] = r < nvalid;
      int arow = a_ok[i] ? (rowmap ? rowmap[rowbase + r] : (rowbase + r)) : arow0;
      a_src[i] = Abase + (size_t)arow * Kdim + kg * 8;
      a_off[i] = r * LDA + kg * 8;
    }
  }
  const float* bf_src[2];          // BT=false: float4 from rows k,k+1
  const unsigned short* bt_src[2]; // BT=true : 8 halves of one column
  int b_off[2];
#pragma unroll
  for (int i = 0; i < 2; ++i) {
    int idx = tid + i * 256;
    if (BT) {
      int col = idx >> 2, kg = idx & 3;
      bt_src[i] = Bt + (size_t)(n0 + col) * Kdim + kg * 8;
      b_off[i]  = col * LDA + kg * 8;
      bf_src[i] = nullptr;
    } else {
      int ng = idx & 31, k = (idx >> 5) * 2;
      bf_src[i] = Bf + (size_t)k * Ndim + n0 + ng * 4;
      b_off[i]  = (ng * 4) * LDA + k;
      bt_src[i] = nullptr;
    }
  }

  uint4  areg[2], btreg[2];
  float4 br0[2], br1[2];

  // ---- prologue: fetch K-tile 0 ------------------------------------------
#pragma unroll
  for (int i = 0; i < 2; ++i) {
    areg[i] = *(const uint4*)(a_src[i]);
    if (BT) {
      btreg[i] = *(const uint4*)(bt_src[i]);
    } else {
      br0[i] = *(const float4*)(bf_src[i]);
      br1[i] = *(const float4*)(bf_src[i] + Ndim);
    }
  }
  int pp = 0;
#pragma unroll
  for (int i = 0; i < 2; ++i) {
    uint4 v = areg[i];
    if (!a_ok[i]) { v.x = v.y = v.z = v.w = 0u; }
    *(uint4*)&ldsA[pp][a_off[i]] = v;
    if (BT) {
      *(uint4*)&ldsB[pp][b_off[i]] = btreg[i];
    } else {
      *(unsigned int*)&ldsB[pp][b_off[i] + 0 * LDA] = f2bf_pk(br0[i].x, br1[i].x);
      *(unsigned int*)&ldsB[pp][b_off[i] + 1 * LDA] = f2bf_pk(br0[i].y, br1[i].y);
      *(unsigned int*)&ldsB[pp][b_off[i] + 2 * LDA] = f2bf_pk(br0[i].z, br1[i].z);
      *(unsigned int*)&ldsB[pp][b_off[i] + 3 * LDA] = f2bf_pk(br0[i].w, br1[i].w);
    }
  }

  for (int kk = 0; kk < Kdim; kk += 32) {
    __syncthreads();                          // stage pp ready for all waves
    const bool more = (kk + 32) < Kdim;
    // ---- issue next tile's global loads (latency hidden under WMMA) ------
    if (more) {
#pragma unroll
      for (int i = 0; i < 2; ++i) {
        areg[i] = *(const uint4*)(a_src[i] + kk + 32);
        if (BT) {
          btreg[i] = *(const uint4*)(bt_src[i] + kk + 32);
        } else {
          br0[i] = *(const float4*)(bf_src[i] + (size_t)(kk + 32) * Ndim);
          br1[i] = *(const float4*)(bf_src[i] + (size_t)(kk + 33) * Ndim);
        }
      }
    }
    // ---- compute: A frag once; B frags ping-pong 2-deep so each WMMA only
    // waits on LDS loads issued one stage earlier --------------------------
    Frag af;
    {
      int arow = wave * 16 + lh;
#pragma unroll
      for (int p = 0; p < 8; ++p) {
        int kp = ((p < 4) ? 2 * p : 8 + 2 * p) + (hi ? 8 : 0);
        af.u[p] = *(const unsigned int*)&ldsA[pp][arow * LDA + kp];
      }
    }
    const int kb = hi ? 8 : 0;
    Frag bb[2];
#pragma unroll
    for (int p = 0; p < 8; ++p)
      bb[0].u[p] = *(const unsigned int*)&ldsB[pp][lh * LDA + kb + ((p < 4) ? 2 * p : 8 + 2 * p)];
#pragma unroll
    for (int n = 0; n < 8; ++n) {
      if (n < 7) {
        int bn = (n + 1) * 16 + lh;
#pragma unroll
        for (int p = 0; p < 8; ++p)
          bb[(n + 1) & 1].u[p] =
              *(const unsigned int*)&ldsB[pp][bn * LDA + kb + ((p < 4) ? 2 * p : 8 + 2 * p)];
      }
      acc[n] = __builtin_amdgcn_wmma_f32_16x16x32_bf16(
          false, af.v, false, bb[n & 1].v, (short)0, acc[n], false, false);
    }
    // ---- store next tile into the other buffer ---------------------------
    if (more) {
      int np = pp ^ 1;
#pragma unroll
      for (int i = 0; i < 2; ++i) {
        uint4 v = areg[i];
        if (!a_ok[i]) { v.x = v.y = v.z = v.w = 0u; }
        *(uint4*)&ldsA[np][a_off[i]] = v;
        if (BT) {
          *(uint4*)&ldsB[np][b_off[i]] = btreg[i];
        } else {
          *(unsigned int*)&ldsB[np][b_off[i] + 0 * LDA] = f2bf_pk(br0[i].x, br1[i].x);
          *(unsigned int*)&ldsB[np][b_off[i] + 1 * LDA] = f2bf_pk(br0[i].y, br1[i].y);
          *(unsigned int*)&ldsB[np][b_off[i] + 2 * LDA] = f2bf_pk(br0[i].z, br1[i].z);
          *(unsigned int*)&ldsB[np][b_off[i] + 3 * LDA] = f2bf_pk(br0[i].w, br1[i].w);
        }
      }
    }
    pp ^= 1;
  }
}

// ---------------------------------------------------------------------------
// Init / conversion kernels
// ---------------------------------------------------------------------------
__global__ void k_zero_out(float* __restrict__ out) {
  size_t i = ((size_t)blockIdx.x * 256 + threadIdx.x) * 4;
  float4 z; z.x = z.y = z.z = z.w = 0.f;
  *(float4*)(out + i) = z;
}
__global__ void k_zero_counts(unsigned int* __restrict__ counts) {
  if (threadIdx.x < NEXP) counts[threadIdx.x] = 0u;
}
__global__ void k_cvt_x(const float* __restrict__ x, unsigned short* __restrict__ xb) {
  size_t i = ((size_t)blockIdx.x * 256 + threadIdx.x) * 4;
  float4 v = *(const float4*)(x + i);
  uint2 p;
  p.x = f2bf_pk(v.x, v.y);
  p.y = f2bf_pk(v.z, v.w);
  *(uint2*)(xb + i) = p;
}
// Transpose+convert: W fp32 [K][N] -> Wt bf16 [N][K]. grid (N/32, K/32).
__global__ void k_cvt_t(const float* __restrict__ W, unsigned short* __restrict__ Wt,
                        int K, int N) {
  __shared__ unsigned short t[32][33];
  int k0 = blockIdx.y * 32, c0 = blockIdx.x * 32;
  int tid = threadIdx.x;
#pragma unroll
  for (int i = 0; i < 4; ++i) {
    int idx = tid + i * 256;
    int r = idx >> 5, c = idx & 31;
    t[c][r] = f2bf(W[(size_t)(k0 + r) * N + c0 + c]);
  }
  __syncthreads();
#pragma unroll
  for (int i = 0; i < 2; ++i) {
    int idx = tid + i * 256;
    int c = idx >> 4, rp = (idx & 15) * 2;
    unsigned int u = (unsigned int)t[c][rp] | ((unsigned int)t[c][rp + 1] << 16);
    *(unsigned int*)&Wt[(size_t)(c0 + c) * K + k0 + rp] = u;
  }
}

// ---------------------------------------------------------------------------
// Router hidden: h = gelu(xb @ wr1 + br1), bf16 out [NT x DHID]
// grid (DHID/128, NT/128), 256 threads
// ---------------------------------------------------------------------------
template <bool BT>
__global__ void k_router_h(const unsigned short* __restrict__ xb,
                           const float* __restrict__ wr1,
                           const unsigned short* __restrict__ wr1t,
                           const float* __restrict__ br1,
                           unsigned short* __restrict__ h) {
  __shared__ unsigned short ldsA[2][128 * LDA];
  __shared__ unsigned short ldsB[2][128 * LDA];
  int n0 = blockIdx.x * 128, m0 = blockIdx.y * 128;
  v8f acc[8] = {};
  mma_tile<BT>(xb, DM, nullptr, m0, 128, wr1, wr1t, DHID, n0, ldsA, ldsB, acc);

  int lane = threadIdx.x & 31, wave = threadIdx.x >> 5;
  int colin = lane & 15; int radd = (lane >= 16) ? 8 : 0;
#pragma unroll
  for (int n = 0; n < 8; ++n) {
    int gn = n0 + n * 16 + colin;
    float bias = br1[gn];
#pragma unroll
    for (int r = 0; r < 8; ++r) {
      int gm = m0 + wave * 16 + r + radd;
      h[(size_t)gm * DHID + gn] = f2bf(gelu_f(acc[n][r] + bias));
    }
  }
}

// ---------------------------------------------------------------------------
// Gate: logits = h @ wr2 + br2; top-2 softmax; build per-expert token lists.
// one wave per token; grid NT/8, 256 threads
// ---------------------------------------------------------------------------
__global__ void k_gate(const unsigned short* __restrict__ h,
                       const float* __restrict__ wr2,
                       const float* __restrict__ br2,
                       float* __restrict__ gate_out,
                       int* __restrict__ list,
                       unsigned int* __restrict__ counts) {
  int wave = threadIdx.x >> 5, lane = threadIdx.x & 31;
  int t = blockIdx.x * 8 + wave;
  const unsigned short* hp = h + (size_t)t * DHID + lane * 16;
  float acc[NEXP] = {0.f, 0.f, 0.f, 0.f};
#pragma unroll
  for (int j = 0; j < 16; ++j) {
    float hv = bf2f(hp[j]);
    const float* w = wr2 + (size_t)(lane * 16 + j) * NEXP;
#pragma unroll
    for (int e = 0; e < NEXP; ++e) acc[e] += hv * w[e];
  }
#pragma unroll
  for (int off = 16; off >= 1; off >>= 1)
#pragma unroll
    for (int e = 0; e < NEXP; ++e) acc[e] += __shfl_xor(acc[e], off, 32);

  if (lane == 0) {
    float lg[NEXP];
#pragma unroll
    for (int e = 0; e < NEXP; ++e) lg[e] = acc[e] + br2[e];
    float m0v = lg[0]; int m0i = 0;
#pragma unroll
    for (int e = 1; e < NEXP; ++e) if (lg[e] > m0v) { m0v = lg[e]; m0i = e; }
    float m1v = -3.4e38f;
#pragma unroll
    for (int e = 0; e < NEXP; ++e) if (e != m0i && lg[e] > m1v) m1v = lg[e];
    float thresh = m1v, s = 0.f, g[NEXP];
#pragma unroll
    for (int e = 0; e < NEXP; ++e) {
      g[e] = (lg[e] >= thresh) ? __expf(lg[e] - m0v) : 0.f;
      s += g[e];
    }
    float inv = 1.f / s;
#pragma unroll
    for (int e = 0; e < NEXP; ++e) {
      g[e] *= inv;
      gate_out[(size_t)t * NEXP + e] = g[e];
      if (g[e] > 0.f) {
        unsigned int pos = atomicAdd(&counts[e], 1u);
        list[(size_t)e * NT + pos] = t;
      }
    }
  }
}

// ---------------------------------------------------------------------------
// Expert FFN1: hid = gelu(gather(xb, list) @ w1_e + b1_e), bf16 [CHUNK x DFF]
// grid (DFF/128, CHUNK/128)
// ---------------------------------------------------------------------------
template <bool BT>
__global__ void k_ffn1(const unsigned short* __restrict__ xb,
                       const float* __restrict__ w1e,
                       const unsigned short* __restrict__ w1te,
                       const float* __restrict__ b1e,
                       const int* __restrict__ list_e,
                       const unsigned int* __restrict__ cnt_e,
                       unsigned short* __restrict__ hid,
                       int chunk) {
  __shared__ unsigned short ldsA[2][128 * LDA];
  __shared__ unsigned short ldsB[2][128 * LDA];
  int count = (int)*cnt_e;
  int n0 = blockIdx.x * 128;
  int mloc = blockIdx.y * 128;
  int rowbase = chunk * CHUNK + mloc;
  if (rowbase >= count) return;
  int nvalid = count - rowbase; if (nvalid > 128) nvalid = 128;

  v8f acc[8] = {};
  mma_tile<BT>(xb, DM, list_e, rowbase, nvalid, w1e, w1te, DFF, n0, ldsA, ldsB, acc);

  int lane = threadIdx.x & 31, wave = threadIdx.x >> 5;
  int colin = lane & 15; int radd = (lane >= 16) ? 8 : 0;
#pragma unroll
  for (int n = 0; n < 8; ++n) {
    int gn = n0 + n * 16 + colin;
    float bias = b1e[gn];
#pragma unroll
    for (int r = 0; r < 8; ++r) {
      int rloc = wave * 16 + r + radd;
      if (rloc < nvalid)
        hid[(size_t)(mloc + rloc) * DFF + gn] = f2bf(gelu_f(acc[n][r] + bias));
    }
  }
}

// ---------------------------------------------------------------------------
// Expert FFN2: out[token] += gate[token][e] * (hid @ w2_e + b2_e)
// grid (DM/128, CHUNK/128). Experts serialized on stream -> plain RMW is safe.
// ---------------------------------------------------------------------------
template <bool BT>
__global__ void k_ffn2(const unsigned short* __restrict__ hid,
                       const float* __restrict__ w2e,
                       const unsigned short* __restrict__ w2te,
                       const float* __restrict__ b2e,
                       const int* __restrict__ list_e,
                       const unsigned int* __restrict__ cnt_e,
                       const float* __restrict__ gate,
                       float* __restrict__ out,
                       int expert, int chunk) {
  __shared__ unsigned short ldsA[2][128 * LDA];
  __shared__ unsigned short ldsB[2][128 * LDA];
  int count = (int)*cnt_e;
  int n0 = blockIdx.x * 128;
  int mloc = blockIdx.y * 128;
  int rowbase = chunk * CHUNK + mloc;
  if (rowbase >= count) return;
  int nvalid = count - rowbase; if (nvalid > 128) nvalid = 128;

  v8f acc[8] = {};
  mma_tile<BT>(hid, DFF, nullptr, mloc, nvalid, w2e, w2te, DM, n0, ldsA, ldsB, acc);

  int lane = threadIdx.x & 31, wave = threadIdx.x >> 5;
  int colin = lane & 15; int radd = (lane >= 16) ? 8 : 0;
#pragma unroll
  for (int n = 0; n < 8; ++n) {
    int gn = n0 + n * 16 + colin;
    float bias = b2e[gn];
#pragma unroll
    for (int r = 0; r < 8; ++r) {
      int rloc = wave * 16 + r + radd;
      if (rloc < nvalid) {
        int token = list_e[rowbase + rloc];
        float g = gate[(size_t)token * NEXP + expert];
        float* po = out + (size_t)token * DM + gn;
        *po += g * (acc[n][r] + bias);
      }
    }
  }
}

// ---------------------------------------------------------------------------
extern "C" void kernel_launch(void* const* d_in, const int* in_sizes, int n_in,
                              void* d_out, int out_size, void* d_ws, size_t ws_size,
                              hipStream_t stream) {
  (void)in_sizes; (void)n_in; (void)out_size;
  const float* x   = (const float*)d_in[0];
  const float* w1  = (const float*)d_in[1];
  const float* b1  = (const float*)d_in[2];
  const float* w2  = (const float*)d_in[3];
  const float* b2  = (const float*)d_in[4];
  const float* wr1 = (const float*)d_in[5];
  const float* br1 = (const float*)d_in[6];
  const float* wr2 = (const float*)d_in[7];
  const float* br2 = (const float*)d_in[8];

  float* out      = (float*)d_out;                       // [NT x DM]
  float* gate_out = (float*)d_out + (size_t)NT * DM;     // [NT x NEXP]

  // ---- ws layout (deterministic) ----
  char* ws = (char*)d_ws;
  size_t off = 0;
  auto take = [&](size_t bytes) { size_t o = off; off += (bytes + 255) & ~(size_t)255; return o; };
  unsigned short* xb   = (unsigned short*)(ws + take((size_t)NT * DM * 2));     // 16 MiB
  unsigned short* h    = (unsigned short*)(ws + take((size_t)NT * DHID * 2));   //  8 MiB
  unsigned short* hid  = (unsigned short*)(ws + take((size_t)CHUNK * DFF * 2)); // 16 MiB
  unsigned int*   counts = (unsigned int*)(ws + take(256));
  int*            list   = (int*)(ws + take((size_t)NEXP * NT * 4));            // 128 KiB
  unsigned short* wr1t = (unsigned short*)(ws + take((size_t)DHID * DM * 2));        //  1 MiB
  unsigned short* w1t  = (unsigned short*)(ws + take((size_t)NEXP * DM * DFF * 2));  // 32 MiB
  unsigned short* w2t  = (unsigned short*)(ws + take((size_t)NEXP * DFF * DM * 2));  // 32 MiB
  const bool pre = ws_size >= off;          // pre-transposed bf16 weights fit?

  k_zero_counts<<<1, 32, 0, stream>>>(counts);
  k_zero_out<<<(NT * DM) / (256 * 4), 256, 0, stream>>>(out);
  k_cvt_x<<<(NT * DM) / (256 * 4), 256, 0, stream>>>(x, xb);

  if (pre) {
    k_cvt_t<<<dim3(DHID / 32, DM / 32), 256, 0, stream>>>(wr1, wr1t, DM, DHID);
    for (int e = 0; e < NEXP; ++e) {
      k_cvt_t<<<dim3(DFF / 32, DM / 32), 256, 0, stream>>>(
          w1 + (size_t)e * DM * DFF, w1t + (size_t)e * DFF * DM, DM, DFF);
      k_cvt_t<<<dim3(DM / 32, DFF / 32), 256, 0, stream>>>(
          w2 + (size_t)e * DFF * DM, w2t + (size_t)e * DM * DFF, DFF, DM);
    }
    k_router_h<true><<<dim3(DHID / 128, NT / 128), 256, 0, stream>>>(
        xb, wr1, wr1t, br1, h);
  } else {
    k_router_h<false><<<dim3(DHID / 128, NT / 128), 256, 0, stream>>>(
        xb, wr1, nullptr, br1, h);
  }

  k_gate<<<NT / 8, 256, 0, stream>>>(h, wr2, br2, gate_out, list, counts);

  for (int e = 0; e < NEXP; ++e) {
    const float* w1e = w1 + (size_t)e * DM * DFF;
    const float* b1e = b1 + (size_t)e * DFF;
    const float* w2e = w2 + (size_t)e * DFF * DM;
    const float* b2e = b2 + (size_t)e * DM;
    const unsigned short* w1te = w1t + (size_t)e * DFF * DM;
    const unsigned short* w2te = w2t + (size_t)e * DM * DFF;
    const int* le = list + (size_t)e * NT;
    const unsigned int* ce = counts + e;
    for (int c = 0; c < NT / CHUNK; ++c) {
      if (pre) {
        k_ffn1<true><<<dim3(DFF / 128, CHUNK / 128), 256, 0, stream>>>(
            xb, w1e, w1te, b1e, le, ce, hid, c);
        k_ffn2<true><<<dim3(DM / 128, CHUNK / 128), 256, 0, stream>>>(
            hid, w2e, w2te, b2e, le, ce, gate_out, out, e, c);
      } else {
        k_ffn1<false><<<dim3(DFF / 128, CHUNK / 128), 256, 0, stream>>>(
            xb, w1e, nullptr, b1e, le, ce, hid, c);
        k_ffn2<false><<<dim3(DM / 128, CHUNK / 128), 256, 0, stream>>>(
            hid, w2e, nullptr, b2e, le, ce, gate_out, out, e, c);
      }
    }
  }
}
